// Model_50422916055256
// MI455X (gfx1250) — matrix operand order repaired and checked by reading, not compiled, not run on hardware
//
#include <hip/hip_runtime.h>
#include <hip/hip_bf16.h>
#include <stdint.h>

#define BATCH  32
#define SEQ    2048
#define PRED   1024
#define PERIOD 256
#define NCH    256
#define KSEG   8
#define LSEG   4

typedef __bf16 bf16;
typedef bf16  v16bf __attribute__((ext_vector_type(16)));
typedef float v8f   __attribute__((ext_vector_type(8)));
typedef int   nint4 __attribute__((vector_size(16)));   // native vector int4 (not HIP_vector_type)

union Frag { v16bf v; int4 q[2]; };

#if defined(__has_builtin)
#if __has_builtin(__builtin_amdgcn_global_load_async_to_lds_b128)
#define USE_ASYNC_LDS 1
#endif
#if __has_builtin(__builtin_amdgcn_s_wait_asynccnt)
#define HAVE_WAIT_ASYNC_BUILTIN 1
#endif
#endif

__device__ __forceinline__ void wait_async0() {
#if defined(HAVE_WAIT_ASYNC_BUILTIN)
  __builtin_amdgcn_s_wait_asynccnt(0);
#else
  asm volatile("s_wait_asynccnt 0" ::: "memory");
#endif
}

__device__ __forceinline__ uint16_t f2bf(float f) {
  union { float f; uint32_t u; } x; x.f = f;
  uint32_t r = x.u + 0x7FFFu + ((x.u >> 16) & 1u);   // round-to-nearest-even
  return (uint16_t)(r >> 16);
}

// ---------------- K1a: partial sums over time --------------------------------
__global__ void stats_partial(const float* __restrict__ x,
                              float* __restrict__ psum, float* __restrict__ psq) {
  int sc = blockIdx.x;            // 0..15 (chunks of 128 time steps)
  int b  = blockIdx.y;            // 0..31
  int n  = threadIdx.x;           // 0..255
  const float* p = x + ((size_t)b * SEQ + (size_t)sc * 128) * NCH + n;
  float s = 0.f, s2 = 0.f;
#pragma unroll 4
  for (int i = 0; i < 128; ++i) { float v = p[(size_t)i * NCH]; s += v; s2 += v * v; }
  int o = (b * 16 + sc) * NCH + n;
  psum[o] = s; psq[o] = s2;
}

// ---------------- K1b: finalize mean / stdev ---------------------------------
__global__ void stats_final(const float* __restrict__ psum, const float* __restrict__ psq,
                            float* __restrict__ meanv, float* __restrict__ stdv) {
  int b = blockIdx.x, n = threadIdx.x;
  float s = 0.f, s2 = 0.f;
#pragma unroll
  for (int sc = 0; sc < 16; ++sc) { int o = (b * 16 + sc) * NCH + n; s += psum[o]; s2 += psq[o]; }
  float m   = s  * (1.0f / SEQ);
  float var = s2 * (1.0f / SEQ) - m * m;
  meanv[b * NCH + n] = m;
  stdv [b * NCH + n] = sqrtf(var + 1e-5f);
}

// ---------------- K2: normalize + transpose to Xt[b*256+n][s] (bf16) ---------
__global__ void build_xt(const float* __restrict__ x,
                         const float* __restrict__ meanv, const float* __restrict__ stdv,
                         uint16_t* __restrict__ xt) {
  __shared__ float tile[64][65];
  int s0 = blockIdx.x * 64;       // 32 tiles over SEQ
  int n0 = blockIdx.y * 64;       // 4 tiles over NCH
  int b  = blockIdx.z;
  int t  = threadIdx.x;
  int nl = t & 63, sl0 = t >> 6;
#pragma unroll
  for (int r = 0; r < 16; ++r) {
    int sl = sl0 + r * 4;
    tile[sl][nl] = x[((size_t)b * SEQ + s0 + sl) * NCH + n0 + nl];
  }
  __syncthreads();
  int sl = t & 63, nl0 = t >> 6;
#pragma unroll
  for (int r = 0; r < 16; ++r) {
    int nn = nl0 + r * 4;
    int col = n0 + nn;
    float m  = meanv[b * NCH + col];
    float sd = stdv [b * NCH + col];
    float v  = (tile[sl][nn] - m) / sd;
    xt[((size_t)(b * NCH + col)) * SEQ + s0 + sl] = f2bf(v);
  }
}

// ---------------- K3: batched 256x256x256 fp32 GEMMs for W powers -------------
struct PowJobs { int a[4]; int b[4]; int c[4]; };

__global__ void pow_gemm(float* __restrict__ pw, PowJobs j) {
  int z = blockIdx.y;
  const float* A = pw + (size_t)j.a[z] * 65536;
  const float* B = pw + (size_t)j.b[z] * 65536;
  float*       C = pw + (size_t)j.c[z] * 65536;
  int q = blockIdx.x, n = threadIdx.x;
  const float* ar = A + q * 256;
  float acc = 0.f;
#pragma unroll 8
  for (int p = 0; p < 256; ++p) acc = fmaf(ar[p], B[p * 256 + n], acc);
  C[q * 256 + n] = acc;
}

// ---------------- K4: A[l*256+q][k*256+p] = W_enc[l,k]*W^(8-k+l) (bf16) -------
__global__ void build_a(const float* __restrict__ wbase, const float* __restrict__ pw,
                        const float* __restrict__ wenc, uint16_t* __restrict__ Abf) {
  int q = blockIdx.x, k = blockIdx.y, l = blockIdx.z, p = threadIdx.x;
  int e = KSEG - k + l;                                   // 1..11
  const float* src = (e == 1) ? wbase : (pw + (size_t)e * 65536);
  float v = wenc[l * KSEG + k] * src[q * 256 + p];
  Abf[((size_t)(l * 256 + q)) * SEQ + k * 256 + p] = f2bf(v);
}

// ---------------- K5: D[1024x8192] = A x X  (WMMA bf16) + fused epilogue ------
// Block tile 128x128, 8 waves of 32; wave tile 64(M) x 32(N); K-step 32.
// LDS tiles row-major, 32 bf16 per row, padded to 80B rows (16B aligned).
// Global->LDS path: GLOBAL_LOAD_ASYNC_TO_LDS_B128 (ASYNCcnt) when available.
__global__ __launch_bounds__(256) void gemm_wmma(
    const uint16_t* __restrict__ Abf, const uint16_t* __restrict__ Xt,
    const float* __restrict__ meanv, const float* __restrict__ stdv,
    const float* __restrict__ benc, float* __restrict__ out) {
  __shared__ int4 smem[2][1280];                 // per buffer: A 10240B + B 10240B

  const int t    = threadIdx.x;
  const int lane = t & 31;
  const int w    = t >> 5;
  const int wm   = w & 1;                        // 2 waves over M
  const int wn   = w >> 1;                       // 4 waves over N
  const int bm   = blockIdx.y;                   // 8 M-tiles
  const int bn   = blockIdx.x;                   // 64 N-tiles

  // global->LDS cooperative loader mapping: thread t -> (row, half-row of 16 bf16)
  const int lrow  = t >> 1;
  const int lhalf = t & 1;
  const uint16_t* gA = Abf + ((size_t)(bm * 128 + lrow)) * SEQ + lhalf * 16;
  const uint16_t* gB = Xt  + ((size_t)(bn * 128 + lrow)) * SEQ + lhalf * 16;

  // per-thread LDS write addresses for the two buffers
  char* wrA[2] = { (char*)smem[0] + lrow * 80 + lhalf * 32,
                   (char*)smem[1] + lrow * 80 + lhalf * 32 };

  v8f acc[4][2];
  const v8f vzero = {0.f, 0.f, 0.f, 0.f, 0.f, 0.f, 0.f, 0.f};
#pragma unroll
  for (int i = 0; i < 4; ++i)
#pragma unroll
    for (int jj = 0; jj < 2; ++jj) acc[i][jj] = vzero;

  // fragment read offsets (contiguous k order k = 16*(lane/16) + s, same for A and B)
  const int arow = lane & 15;
  const int aoff = (lane < 16) ? 0 : 32;         // A: lanes<16 K{0-15};      else K{16-31}
  const int boff = (lane < 16) ? 0 : 32;         // B: lanes<16 K{0-15};      else K{16-31}

#if defined(USE_ASYNC_LDS)
  typedef __attribute__((address_space(1))) nint4* gp1;  // global src: native int4, addrspace(1)
  typedef __attribute__((address_space(3))) nint4* lp3;  // LDS dst:    native int4, addrspace(3)
  // kick off K-tile 0 into buffer 0 (INST_OFFSET applies to both global and LDS sides)
  {
    char* dA = wrA[0];
    __builtin_amdgcn_global_load_async_to_lds_b128((gp1)gA, (lp3)dA,            0, 0);
    __builtin_amdgcn_global_load_async_to_lds_b128((gp1)gA, (lp3)dA,           16, 0);
    __builtin_amdgcn_global_load_async_to_lds_b128((gp1)gB, (lp3)(dA + 10240),  0, 0);
    __builtin_amdgcn_global_load_async_to_lds_b128((gp1)gB, (lp3)(dA + 10240), 16, 0);
  }
  wait_async0();
  __syncthreads();

  for (int it = 0; it < 64; ++it) {
    const int cur = it & 1;
    if (it < 63) {                               // async-stage next K-tile into other buffer
      const int kk = (it + 1) * 32;
      char* dA = wrA[cur ^ 1];
      __builtin_amdgcn_global_load_async_to_lds_b128((gp1)(gA + kk), (lp3)dA,            0, 0);
      __builtin_amdgcn_global_load_async_to_lds_b128((gp1)(gA + kk), (lp3)dA,           16, 0);
      __builtin_amdgcn_global_load_async_to_lds_b128((gp1)(gB + kk), (lp3)(dA + 10240),  0, 0);
      __builtin_amdgcn_global_load_async_to_lds_b128((gp1)(gB + kk), (lp3)(dA + 10240), 16, 0);
    }

    char* cb = (char*)smem[cur];
    Frag af[4];
#pragma unroll
    for (int i = 0; i < 4; ++i) {
      char* p = cb + (wm * 64 + i * 16 + arow) * 80 + aoff;
      af[i].q[0] = *(const int4*)p;
      af[i].q[1] = *(const int4*)(p + 16);
    }
    Frag bfr[2];
#pragma unroll
    for (int jj = 0; jj < 2; ++jj) {
      char* p = cb + 10240 + (wn * 32 + jj * 16 + arow) * 80 + boff;
      bfr[jj].q[0] = *(const int4*)p;
      bfr[jj].q[1] = *(const int4*)(p + 16);
    }
#pragma unroll
    for (int i = 0; i < 4; ++i)
#pragma unroll
      for (int jj = 0; jj < 2; ++jj)
        acc[i][jj] = __builtin_amdgcn_wmma_f32_16x16x32_bf16(
            false, af[i].v, false, bfr[jj].v, (short)0, acc[i][jj], false, false);

    if (it < 63) wait_async0();                  // our LDS writes landed
    __syncthreads();                             // everyone's writes visible
  }
#else
  // -------- fallback: register-staged global->LDS pipeline --------
  int4 ra0, ra1, rb0, rb1;
  {
    const int4* pa = (const int4*)(gA);
    const int4* pb = (const int4*)(gB);
    ra0 = pa[0]; ra1 = pa[1]; rb0 = pb[0]; rb1 = pb[1];
    int4* dA = (int4*)wrA[0];
    dA[0] = ra0; dA[1] = ra1;
    int4* dB = (int4*)(wrA[0] + 10240);
    dB[0] = rb0; dB[1] = rb1;
  }
  __syncthreads();

  for (int it = 0; it < 64; ++it) {
    const int cur = it & 1;
    if (it < 63) {
      const int kk = (it + 1) * 32;
      const int4* pa = (const int4*)(gA + kk);
      const int4* pb = (const int4*)(gB + kk);
      ra0 = pa[0]; ra1 = pa[1]; rb0 = pb[0]; rb1 = pb[1];
    }

    char* cb = (char*)smem[cur];
    Frag af[4];
#pragma unroll
    for (int i = 0; i < 4; ++i) {
      char* p = cb + (wm * 64 + i * 16 + arow) * 80 + aoff;
      af[i].q[0] = *(const int4*)p;
      af[i].q[1] = *(const int4*)(p + 16);
    }
    Frag bfr[2];
#pragma unroll
    for (int jj = 0; jj < 2; ++jj) {
      char* p = cb + 10240 + (wn * 32 + jj * 16 + arow) * 80 + boff;
      bfr[jj].q[0] = *(const int4*)p;
      bfr[jj].q[1] = *(const int4*)(p + 16);
    }
#pragma unroll
    for (int i = 0; i < 4; ++i)
#pragma unroll
      for (int jj = 0; jj < 2; ++jj)
        acc[i][jj] = __builtin_amdgcn_wmma_f32_16x16x32_bf16(
            false, af[i].v, false, bfr[jj].v, (short)0, acc[i][jj], false, false);

    if (it < 63) {
      int4* dA = (int4*)wrA[cur ^ 1];
      dA[0] = ra0; dA[1] = ra1;
      int4* dB = (int4*)(wrA[cur ^ 1] + 10240);
      dB[0] = rb0; dB[1] = rb1;
    }
    __syncthreads();
  }
#endif

  // epilogue: out[b,t,n] = (acc + b_enc[t/256]) * stdev[b,n] + mean[b,n]
#pragma unroll
  for (int i = 0; i < 4; ++i) {
    const int mBase = bm * 128 + wm * 64 + i * 16;
    const float bias = benc[mBase >> 8];
    const int r0 = mBase + ((lane < 16) ? 0 : 8);
#pragma unroll
    for (int jj = 0; jj < 2; ++jj) {
      const int col = bn * 128 + wn * 32 + jj * 16 + (lane & 15);   // b*256+n
      const float sd = stdv[col];
      const float mu = meanv[col];
      const int bb = col >> 8, nn = col & 255;
      float* po = out + (size_t)bb * (PRED * NCH) + nn;
#pragma unroll
      for (int r = 0; r < 8; ++r)
        po[(size_t)(r0 + r) * NCH] = (acc[i][jj][r] + bias) * sd + mu;
    }
  }
}

// -----------------------------------------------------------------------------
extern "C" void kernel_launch(void* const* d_in, const int* in_sizes, int n_in,
                              void* d_out, int out_size, void* d_ws, size_t ws_size,
                              hipStream_t stream) {
  const float* x_enc  = (const float*)d_in[0];
  const float* W_base = (const float*)d_in[4];
  const float* W_enc  = (const float*)d_in[5];
  const float* b_enc  = (const float*)d_in[6];
  float* out = (float*)d_out;

  // workspace carve-up (all 32B aligned)
  float* psum   = (float*)d_ws;                          // 32*16*256
  float* psq    = psum + 32 * 16 * 256;                  // 32*16*256
  float* meanv  = psq  + 32 * 16 * 256;                  // 8192
  float* stdv   = meanv + 8192;                          // 8192
  float* powers = stdv + 8192;                           // 12 * 65536 (slot e = W^e)
  uint16_t* Abf = (uint16_t*)(powers + 12 * 65536);      // 1024*2048 bf16
  uint16_t* Xt  = Abf + (size_t)1024 * 2048;             // 8192*2048 bf16

  stats_partial<<<dim3(16, 32), 256, 0, stream>>>(x_enc, psum, psq);
  stats_final  <<<32, 256, 0, stream>>>(psum, psq, meanv, stdv);
  build_xt     <<<dim3(32, 4, 32), 256, 0, stream>>>(x_enc, meanv, stdv, Xt);

  // power chain: slot1 = W, then log-depth batched products
  (void)hipMemcpyAsync(powers + 65536, W_base, 65536 * sizeof(float),
                       hipMemcpyDeviceToDevice, stream);
  { PowJobs j = {{1, 0, 0, 0}, {1, 0, 0, 0}, {2, 0, 0, 0}};
    pow_gemm<<<dim3(256, 1), 256, 0, stream>>>(powers, j); }
  { PowJobs j = {{1, 2, 0, 0}, {2, 2, 0, 0}, {3, 4, 0, 0}};
    pow_gemm<<<dim3(256, 2), 256, 0, stream>>>(powers, j); }
  { PowJobs j = {{1, 2, 3, 4}, {4, 4, 4, 4}, {5, 6, 7, 8}};
    pow_gemm<<<dim3(256, 4), 256, 0, stream>>>(powers, j); }
  { PowJobs j = {{1, 2, 3, 0}, {8, 8, 8, 0}, {9, 10, 11, 0}};
    pow_gemm<<<dim3(256, 3), 256, 0, stream>>>(powers, j); }

  build_a<<<dim3(256, 8, 4), 256, 0, stream>>>(W_base, powers, W_enc, Abf);

  gemm_wmma<<<dim3(64, 8), 256, 0, stream>>>(Abf, Xt, meanv, stdv, b_enc, out);
}
